// SpatioTemporalFusion_15822659518639
// MI455X (gfx1250) — compile-verified, matrix operand order checked by reading, not run on hardware
//
#include <hip/hip_runtime.h>
#include <hip/hip_bf16.h>
#include <math.h>

typedef __attribute__((ext_vector_type(16))) _Float16 v16h;
typedef __attribute__((ext_vector_type(8)))  _Float16 v8h;
typedef __attribute__((ext_vector_type(8)))  float    v8f;

// ---------------------------------------------------------------------------
// Fragment loader: 16-bit A/B matrix 16x32 (MxK) per CDNA5 WMMA VGPR layout.
// lane 0-15 : row = lane,    K = {0..7} then {16..23}
// lane 16-31: row = lane-16, K = {8..15} then {24..31}
// Two 16B LDS loads per lane; all offsets 16-byte aligned.
// ---------------------------------------------------------------------------
__device__ inline v16h load_frag16(const _Float16* base, int row_stride, int lane) {
    int r  = lane & 15;
    int kb = (lane >> 4) << 3;   // 0 or 8
    const _Float16* p = base + r * row_stride + kb;
    v8h lo = *(const v8h*)(p);
    v8h hi = *(const v8h*)(p + 16);
    return __builtin_shufflevector(lo, hi, 0,1,2,3,4,5,6,7,8,9,10,11,12,13,14,15);
}

__device__ inline void store_tile(const v8f& acc, float* __restrict__ C,
                                  const float* __restrict__ bias,
                                  int m0, int n0, int M, int N, int ldc,
                                  int relu, int lane) {
    int col  = n0 + (lane & 15);
    int half = lane >> 4;
    if (col < N) {
        float bv = bias ? bias[col] : 0.f;
#pragma unroll
        for (int i = 0; i < 8; ++i) {
            int row = m0 + i + 8 * half;
            if (row < M) {
                float v = acc[i] + bv;
                if (relu) v = fmaxf(v, 0.f);
                C[(size_t)row * ldc + col] = v;
            }
        }
    }
}

// ---------------------------------------------------------------------------
// WMMA GEMM: C[M,N] = A[M,K] @ W[N,K]^T + bias, optional ReLU.
// block = 256 threads (8 waves, 2 along M x 4 along N).
// Block tile: 64 rows x 128 cols; each wave owns a 32x32 sub-tile = 2x2
// 16x16 WMMA tiles -> 4 v_wmma per K-chunk with 2x fragment reuse.
// Out-of-range staging rows are CLAMPED (not zero-filled): garbage rows only
// feed output rows/cols >= M/N which the store epilogue guards, so the K-loop
// stays branch-free.
// ---------------------------------------------------------------------------
__global__ __launch_bounds__(256) void stf_gemm_kernel(
    const float* __restrict__ A, const float* __restrict__ W,
    const float* __restrict__ bias, float* __restrict__ C,
    int M, int N, int K, int relu)
{
    __shared__ _Float16 sA[64 * 32];
    __shared__ _Float16 sB[128 * 32];

    const int tid   = threadIdx.x;
    const int lane  = tid & 31;
    const int wave  = tid >> 5;
    const int waveM = wave >> 2;     // 0..1
    const int waveN = wave & 3;      // 0..3
    const int m0     = blockIdx.y * 64;
    const int nblock = blockIdx.x * 128;

    // Staging coordinates (constant across K loop); clamp rows in range.
    const int ar = tid >> 2;               // A row within tile (0..63)
    const int ac = (tid & 3) * 8;          // A col start (0,8,16,24)
    const int br = tid >> 1;               // B row within tile (0..127)
    const int bc = (tid & 1) * 16;         // B col start (0,16)
    const int agr = min(m0 + ar, M - 1);
    const int bgn = min(nblock + br, N - 1);
    _Float16* adst = sA + ar * 32 + ac;
    _Float16* bdst = sB + br * 32 + bc;
    const float* asrc = A + (size_t)agr * K + ac;
    const float* bsrc = W + (size_t)bgn * K + bc;

    v8f acc00 = {}, acc01 = {}, acc10 = {}, acc11 = {};

    for (int kk = 0; kk < K; kk += 32) {
        // Stage A tile: 64x32 f32 -> f16 (8 elements / thread, 2x float4)
        {
            const float4* src = (const float4*)(asrc + kk);
            float4 f0 = src[0], f1 = src[1];
            adst[0] = (_Float16)f0.x; adst[1] = (_Float16)f0.y;
            adst[2] = (_Float16)f0.z; adst[3] = (_Float16)f0.w;
            adst[4] = (_Float16)f1.x; adst[5] = (_Float16)f1.y;
            adst[6] = (_Float16)f1.z; adst[7] = (_Float16)f1.w;
        }
        // Stage B tile: 128x32 f32 -> f16 (16 elements / thread, 4x float4)
        {
            const float4* src = (const float4*)(bsrc + kk);
#pragma unroll
            for (int j = 0; j < 4; ++j) {
                float4 f = src[j];
                bdst[j * 4 + 0] = (_Float16)f.x;
                bdst[j * 4 + 1] = (_Float16)f.y;
                bdst[j * 4 + 2] = (_Float16)f.z;
                bdst[j * 4 + 3] = (_Float16)f.w;
            }
        }

        // Prefetch next K-chunk (global_prefetch_b8)
        if (kk + 32 < K) {
            __builtin_prefetch(asrc + kk + 32, 0, 1);
            __builtin_prefetch(bsrc + kk + 32, 0, 1);
        }

        __syncthreads();

        v16h a0 = load_frag16(sA + (waveM * 32 +  0) * 32, 32, lane);
        v16h a1 = load_frag16(sA + (waveM * 32 + 16) * 32, 32, lane);
        v16h b0 = load_frag16(sB + (waveN * 32 +  0) * 32, 32, lane);
        v16h b1 = load_frag16(sB + (waveN * 32 + 16) * 32, 32, lane);

        acc00 = __builtin_amdgcn_wmma_f32_16x16x32_f16(false, a0, false, b0, (short)0, acc00, false, false);
        acc01 = __builtin_amdgcn_wmma_f32_16x16x32_f16(false, a0, false, b1, (short)0, acc01, false, false);
        acc10 = __builtin_amdgcn_wmma_f32_16x16x32_f16(false, a1, false, b0, (short)0, acc10, false, false);
        acc11 = __builtin_amdgcn_wmma_f32_16x16x32_f16(false, a1, false, b1, (short)0, acc11, false, false);

        __syncthreads();
    }

    const int wm = m0 + waveM * 32;
    const int wn = nblock + waveN * 32;
    store_tile(acc00, C, bias, wm +  0, wn +  0, M, N, N, relu, lane);
    store_tile(acc01, C, bias, wm +  0, wn + 16, M, N, N, relu, lane);
    store_tile(acc10, C, bias, wm + 16, wn +  0, M, N, N, relu, lane);
    store_tile(acc11, C, bias, wm + 16, wn + 16, M, N, N, relu, lane);
}

// ---------------------------------------------------------------------------
// Attention kernel: one block per (b, head). N=512 rows, S=96 keys, HD=32.
// logits = (q*scale) @ k^T via WMMA (K=32, 6 S-tiles); softmax in LDS;
// ctx = P @ V via WMMA (K=96 in 3 chunks, 2 HD-tiles) with transposed V.
// 8 waves each process 4 of the 32 16-row chunks (uniform loop counts).
// ---------------------------------------------------------------------------
__global__ __launch_bounds__(256) void stf_attn_kernel(
    const float* __restrict__ Q, const float* __restrict__ Kf,
    const float* __restrict__ Vf, float* __restrict__ Ctx)
{
    const int b = blockIdx.x >> 3;   // NH = 8
    const int h = blockIdx.x & 7;
    const float* qb = Q   + (size_t)b * 512 * 256 + h * 32;
    const float* kp = Kf  + (size_t)b *  96 * 256 + h * 32;
    const float* vp = Vf  + (size_t)b *  96 * 256 + h * 32;
    float*       cb = Ctx + (size_t)b * 512 * 256 + h * 32;

    __shared__ _Float16 sK [96 * 32];     // k rows   [s][d]
    __shared__ _Float16 sVT[32 * 96];     // v transposed [d][s]
    __shared__ _Float16 sQ [8][16 * 32];  // per-wave q tile
    __shared__ float    sP [8][16 * 96];  // per-wave logits / probs (f32)
    __shared__ _Float16 sPh[8][16 * 96];  // per-wave probs (f16)

    const int tid  = threadIdx.x;
    const int lane = tid & 31;
    const int wave = tid >> 5;

    // Stage K and V^T (3072 elements each, 12 per thread)
    for (int i = tid; i < 96 * 32; i += 256) {
        int s = i >> 5, d = i & 31;
        sK [i]          = (_Float16)kp[(size_t)s * 256 + d];
        sVT[d * 96 + s] = (_Float16)vp[(size_t)s * 256 + d];
    }
    __syncthreads();

    const float scale = 0.17677669529663687f;   // 1/sqrt(32)

    for (int chunk = 0; chunk < 4; ++chunk) {
        const int m0 = (wave * 4 + chunk) * 16;

        // Stage this wave's 16x32 q tile (pre-scaled), 16 halfs per lane
        for (int i = lane; i < 16 * 32; i += 32) {
            int r = i >> 5, d = i & 31;
            sQ[wave][i] = (_Float16)(qb[(size_t)(m0 + r) * 256 + d] * scale);
        }
        __syncthreads();

        v16h qf = load_frag16(sQ[wave], 32, lane);

        // logits: 6 tiles of 16 keys
#pragma unroll
        for (int nt = 0; nt < 6; ++nt) {
            v8f c = {};
            v16h kf = load_frag16(sK + nt * 16 * 32, 32, lane);
            c = __builtin_amdgcn_wmma_f32_16x16x32_f16(
                false, qf, false, kf, (short)0, c, false, false);
            int col  = nt * 16 + (lane & 15);
            int half = lane >> 4;
#pragma unroll
            for (int i = 0; i < 8; ++i)
                sP[wave][(i + 8 * half) * 96 + col] = c[i];
        }
        __syncthreads();

        // Softmax: lanes 0..15 each own one row of 96
        if (lane < 16) {
            float* row = sP[wave] + lane * 96;
            float mx = row[0];
            for (int j = 1; j < 96; ++j) mx = fmaxf(mx, row[j]);
            float sum = 0.f;
            for (int j = 0; j < 96; ++j) { float e = __expf(row[j] - mx); sum += e; row[j] = e; }
            float inv = 1.f / sum;
            for (int j = 0; j < 96; ++j)
                sPh[wave][lane * 96 + j] = (_Float16)(row[j] * inv);
        }
        __syncthreads();

        // ctx = P(16x96) @ V(96x32): 2 HD-tiles, 3 K-chunks of 32
#pragma unroll
        for (int nt = 0; nt < 2; ++nt) {
            v8f c = {};
#pragma unroll
            for (int kc = 0; kc < 3; ++kc) {
                v16h pf = load_frag16(sPh[wave] + kc * 32, 96, lane);
                v16h vf = load_frag16(sVT + nt * 16 * 96 + kc * 32, 96, lane);
                c = __builtin_amdgcn_wmma_f32_16x16x32_f16(
                    false, pf, false, vf, (short)0, c, false, false);
            }
            int col  = nt * 16 + (lane & 15);
            int half = lane >> 4;
#pragma unroll
            for (int i = 0; i < 8; ++i)
                cb[(size_t)(m0 + i + 8 * half) * 256 + col] = c[i];
        }
        __syncthreads();
    }
}

// ---------------------------------------------------------------------------
// Broadcast y (B,N,OUT) -> out (B,S,N,OUT)
// ---------------------------------------------------------------------------
__global__ void stf_bcast_kernel(const float* __restrict__ y, float* __restrict__ out, int total)
{
    int i = blockIdx.x * 256 + threadIdx.x;
    if (i < total) {
        int o  = i & 3;
        int t  = i >> 2;
        int n  = t & 511;           // N = 512
        int t2 = t >> 9;            // b*96 + s
        int b_ = t2 / 96;
        out[i] = y[((size_t)b_ * 512 + n) * 4 + o];
    }
}

extern "C" void kernel_launch(void* const* d_in, const int* in_sizes, int n_in,
                              void* d_out, int out_size, void* d_ws, size_t ws_size,
                              hipStream_t stream) {
    const float* spatial  = (const float*)d_in[0];   // (8,512,256)
    const float* temporal = (const float*)d_in[1];   // (8,96,256)
    const float* Ws  = (const float*)d_in[2];
    const float* bs  = (const float*)d_in[3];
    const float* Wt  = (const float*)d_in[4];
    const float* bt  = (const float*)d_in[5];
    const float* Win = (const float*)d_in[6];        // (768,256)
    const float* bin = (const float*)d_in[7];        // (768,)
    const float* Wao = (const float*)d_in[8];
    const float* bao = (const float*)d_in[9];
    const float* W1  = (const float*)d_in[10];
    const float* b1  = (const float*)d_in[11];
    const float* Wo1 = (const float*)d_in[12];       // (128,256)
    const float* bo1 = (const float*)d_in[13];
    const float* Wo2 = (const float*)d_in[14];       // (4,128)
    const float* bo2 = (const float*)d_in[15];
    float* out = (float*)d_out;

    // Workspace layout (floats), with reuse:
    float* ws   = (float*)d_ws;
    float* sp   = ws;               // 4096x256  -> later "attended"
    float* qbuf = ws + 1048576;     // 4096x256  -> later "fused"
    float* ctx  = ws + 2097152;     // 4096x256  -> later "h2"
    float* tp   = ws + 3145728;     //  768x256  -> later "y"
    float* kbuf = ws + 3342336;     //  768x256
    float* vbuf = ws + 3538944;     //  768x256

    dim3 blk(256);
    dim3 gBig(2, 64);    // N=256, M=4096 (64-row block tiles)
    dim3 gTmp(2, 12);    // N=256, M=768

    // sp = spatial @ Ws^T + bs
    stf_gemm_kernel<<<gBig, blk, 0, stream>>>(spatial, Ws, bs, sp, 4096, 256, 256, 0);
    // tp = temporal @ Wt^T + bt
    stf_gemm_kernel<<<gTmp, blk, 0, stream>>>(temporal, Wt, bt, tp, 768, 256, 256, 0);
    // q = sp @ Wq^T + bq
    stf_gemm_kernel<<<gBig, blk, 0, stream>>>(sp, Win, bin, qbuf, 4096, 256, 256, 0);
    // k = tp @ Wk^T + bk
    stf_gemm_kernel<<<gTmp, blk, 0, stream>>>(tp, Win + 65536,  bin + 256, kbuf, 768, 256, 256, 0);
    // v = tp @ Wv^T + bv
    stf_gemm_kernel<<<gTmp, blk, 0, stream>>>(tp, Win + 131072, bin + 512, vbuf, 768, 256, 256, 0);
    // attention -> ctx
    stf_attn_kernel<<<dim3(64), blk, 0, stream>>>(qbuf, kbuf, vbuf, ctx);
    // attended = ctx @ Wao^T + bao   (reuse sp)
    stf_gemm_kernel<<<gBig, blk, 0, stream>>>(ctx, Wao, bao, sp, 4096, 256, 256, 0);
    // fused = relu(attended @ W1^T + b1)   (reuse qbuf)
    stf_gemm_kernel<<<gBig, blk, 0, stream>>>(sp, W1, b1, qbuf, 4096, 256, 256, 1);
    // h2 = relu(fused @ Wo1^T + bo1)  N=128   (reuse ctx)
    stf_gemm_kernel<<<dim3(1, 64), blk, 0, stream>>>(qbuf, Wo1, bo1, ctx, 4096, 128, 256, 1);
    // y = h2 @ Wo2^T + bo2  N=4, K=128   (reuse tp)
    stf_gemm_kernel<<<dim3(1, 64), blk, 0, stream>>>(ctx, Wo2, bo2, tp, 4096, 4, 128, 0);
    // broadcast (B,N,4) -> (B,S,N,4)
    int total = 8 * 96 * 512 * 4;
    stf_bcast_kernel<<<dim3((total + 255) / 256), blk, 0, stream>>>(tp, out, total);
}
